// HNet_13331578486934
// MI455X (gfx1250) — compile-verified, matrix operand order checked by reading, not run on hardware
//
#include <hip/hip_runtime.h>
#include <hip/hip_bf16.h>

typedef __attribute__((ext_vector_type(16))) __bf16 v16bf;
typedef __attribute__((ext_vector_type(8)))  float  v8f;
typedef __attribute__((ext_vector_type(4)))  unsigned int v4u;
typedef __attribute__((ext_vector_type(4)))  float  v4f;

union Frag { v4u u[2]; v16bf v; };

#define BM 128
#define BN 128
#define BK 32
#define LDA 40   // padded LDS row (bf16 elems) -> 80B stride, conflict-free b128 frag loads

// ---------------------------------------------------------------------------
// Async global -> LDS (CDNA5 GLOBAL_LOAD_ASYNC_TO_LDS_B128, ASYNCcnt-tracked)
// GVS mode: mem_addr = SGPR_base + VGPR_i32_offset
// ---------------------------------------------------------------------------
__device__ __forceinline__ void async_b128(unsigned lds_off, const void* base, int voff_bytes)
{
    asm volatile("global_load_async_to_lds_b128 %0, %1, %2"
                 :: "v"(lds_off), "v"(voff_bytes), "s"(base)
                 : "memory");
}

template<int N>
__device__ __forceinline__ void wait_async()
{
#if __has_builtin(__builtin_amdgcn_s_wait_asynccnt)
    __builtin_amdgcn_s_wait_asynccnt(N);
#else
    asm volatile("s_wait_asynccnt %0" :: "i"(N) : "memory");
#endif
}

// ---------------------------------------------------------------------------
// Transpose + fp32->bf16 convert:  W[K][N] (f32) -> Wt[N][K] (bf16)
// ---------------------------------------------------------------------------
__global__ __launch_bounds__(256)
void transpose_cvt_k(const float* __restrict__ W, __bf16* __restrict__ Wt, int K, int N)
{
    __shared__ __bf16 tile[64 * 72];
    const int k0 = blockIdx.x * 64, n0 = blockIdx.y * 64;
    const int tid = threadIdx.x;
    for (int it = 0; it < 4; ++it) {
        int idx = tid + it * 256;          // 0..1023
        int kl  = idx >> 4;                // 0..63
        int nl  = (idx & 15) * 4;          // 0..60
        v4f v = *(const v4f*)&W[(size_t)(k0 + kl) * N + n0 + nl];
        for (int c = 0; c < 4; ++c)
            tile[(nl + c) * 72 + kl] = (__bf16)v[c];
    }
    __syncthreads();
    for (int it = 0; it < 2; ++it) {
        int idx = tid + it * 256;          // 0..511
        int nl  = idx >> 3;                // 0..63
        int kc  = (idx & 7) * 8;           // 0..56
        *(v4u*)&Wt[(size_t)(n0 + nl) * K + k0 + kc] = *(const v4u*)&tile[nl * 72 + kc];
    }
}

// ---------------------------------------------------------------------------
// RMSNorm row kernel: writes x (bf16) and rmsnorm(x)*w (bf16)
// ---------------------------------------------------------------------------
__global__ __launch_bounds__(256)
void rmsnorm_k(const float* __restrict__ x, const float* __restrict__ w,
               __bf16* __restrict__ xbf, __bf16* __restrict__ xnbf, int d)
{
    __shared__ float red[256];
    const int t = blockIdx.x;
    const float* xr = x + (size_t)t * d;
    float ss = 0.f;
    for (int i = threadIdx.x; i < d; i += 256) { float v = xr[i]; ss += v * v; }
    red[threadIdx.x] = ss; __syncthreads();
    for (int st = 128; st > 0; st >>= 1) {
        if (threadIdx.x < st) red[threadIdx.x] += red[threadIdx.x + st];
        __syncthreads();
    }
    const float scale = rsqrtf(red[0] / (float)d + 1e-5f);
    for (int i = threadIdx.x; i < d; i += 256) {
        float v = xr[i];
        xbf [(size_t)t * d + i] = (__bf16)v;
        xnbf[(size_t)t * d + i] = (__bf16)(v * scale * w[i]);
    }
}

// ---------------------------------------------------------------------------
// bf16 WMMA GEMM: C[M,N](f32) = A[M,K](bf16) x Bt[N,K](bf16)^T  (+ optional residual)
// 128x128x32 tiles, double-buffered LDS filled by async-to-LDS, 8 waves (2Mx4N)
// ---------------------------------------------------------------------------
template<int EPI>   // 0 = plain f32 store, 1 = += residual
__global__ __launch_bounds__(256)
void gemm_bf16_k(const __bf16* __restrict__ A, const __bf16* __restrict__ Bt,
                 float* __restrict__ C, const float* __restrict__ res,
                 int M, int N, int K)
{
    __shared__ __bf16 As[2][BM * LDA];
    __shared__ __bf16 Bs[2][BN * LDA];
    const int tid  = threadIdx.x;
    const int lane = tid & 31, wave = tid >> 5;
    const int waveM = wave >> 2, waveN = wave & 3;
    const int m0 = blockIdx.y * BM, n0 = blockIdx.x * BN;

    const unsigned asB = (unsigned)(size_t)(void*)&As[0][0];
    const unsigned bsB = (unsigned)(size_t)(void*)&Bs[0][0];
    const unsigned bufBytes = BM * LDA * 2;

    const int lrow = tid >> 2;        // 0..63
    const int lch  = (tid & 3) * 8;   // 0,8,16,24

    auto issue = [&](int buf, int kb) {
        for (int it = 0; it < 2; ++it) {
            int r = lrow + it * 64;
            unsigned lo = (unsigned)((r * LDA + lch) * 2) + (unsigned)buf * bufBytes;
            async_b128(asB + lo, A,  ((m0 + r) * K + kb + lch) * 2);
            async_b128(bsB + lo, Bt, ((n0 + r) * K + kb + lch) * 2);
        }
    };

    v8f acc[4][2] = {};
    issue(0, 0);
    const int nsteps = K / BK;
    for (int s = 0; s < nsteps; ++s) {
        const int cur = s & 1;
        if (s + 1 < nsteps) { issue(1 - cur, (s + 1) * BK); wait_async<4>(); }
        else                { wait_async<0>(); }
        __syncthreads();
        const __bf16* as = As[cur];
        const __bf16* bs = Bs[cur];
        const int kl = (lane < 16) ? 0 : 8;   // ISA 16-bit A/B lane K layout
        Frag a[4], b[2];
        for (int i = 0; i < 4; ++i) {
            int m = waveM * 64 + i * 16 + (lane & 15);
            a[i].u[0] = *(const v4u*)&as[m * LDA + kl];
            a[i].u[1] = *(const v4u*)&as[m * LDA + kl + 16];
        }
        for (int j = 0; j < 2; ++j) {
            int n = waveN * 32 + j * 16 + (lane & 15);
            b[j].u[0] = *(const v4u*)&bs[n * LDA + kl];
            b[j].u[1] = *(const v4u*)&bs[n * LDA + kl + 16];
        }
        for (int i = 0; i < 4; ++i)
            for (int j = 0; j < 2; ++j)
                acc[i][j] = __builtin_amdgcn_wmma_f32_16x16x32_bf16(
                    false, a[i].v, false, b[j].v, (short)0, acc[i][j], false, false);
        __syncthreads();
    }

    const int halfsel = (lane >> 4) * 8;
    for (int i = 0; i < 4; ++i)
        for (int j = 0; j < 2; ++j) {
            int n = n0 + waveN * 32 + j * 16 + (lane & 15);
            for (int v = 0; v < 8; ++v) {
                int m = m0 + waveM * 64 + i * 16 + halfsel + v;
                float val = acc[i][j][v];
                if (EPI == 1) val += res[(size_t)m * N + n];
                C[(size_t)m * N + n] = val;
            }
        }
}

// ---------------------------------------------------------------------------
// Fused SwiGLU GEMM: Y[M,N](bf16) = silu(A x BtG^T) * (A x BtH^T)
// Shares the async A stage between the hh and g tiles; double-buffered.
// ---------------------------------------------------------------------------
__global__ __launch_bounds__(256)
void gemm_swiglu_k(const __bf16* __restrict__ A,
                   const __bf16* __restrict__ BtH, const __bf16* __restrict__ BtG,
                   __bf16* __restrict__ Y, int M, int N, int K)
{
    __shared__ __bf16 As [2][BM * LDA];
    __shared__ __bf16 BsH[2][BN * LDA];
    __shared__ __bf16 BsG[2][BN * LDA];
    const int tid  = threadIdx.x;
    const int lane = tid & 31, wave = tid >> 5;
    const int waveM = wave >> 2, waveN = wave & 3;
    const int m0 = blockIdx.y * BM, n0 = blockIdx.x * BN;

    const unsigned asB = (unsigned)(size_t)(void*)&As [0][0];
    const unsigned bhB = (unsigned)(size_t)(void*)&BsH[0][0];
    const unsigned bgB = (unsigned)(size_t)(void*)&BsG[0][0];
    const unsigned bufBytes = BM * LDA * 2;

    const int lrow = tid >> 2;
    const int lch  = (tid & 3) * 8;

    auto issue = [&](int buf, int kb) {
        for (int it = 0; it < 2; ++it) {
            int r = lrow + it * 64;
            unsigned lo = (unsigned)((r * LDA + lch) * 2) + (unsigned)buf * bufBytes;
            int go = ((n0 + r) * K + kb + lch) * 2;
            async_b128(asB + lo, A,   ((m0 + r) * K + kb + lch) * 2);
            async_b128(bhB + lo, BtH, go);
            async_b128(bgB + lo, BtG, go);
        }
    };

    v8f accH[4][2] = {};
    v8f accG[4][2] = {};
    issue(0, 0);
    const int nsteps = K / BK;
    for (int s = 0; s < nsteps; ++s) {
        const int cur = s & 1;
        if (s + 1 < nsteps) { issue(1 - cur, (s + 1) * BK); wait_async<6>(); }
        else                { wait_async<0>(); }
        __syncthreads();
        const __bf16* as = As [cur];
        const __bf16* bh = BsH[cur];
        const __bf16* bg = BsG[cur];
        const int kl = (lane < 16) ? 0 : 8;
        Frag a[4], b[2];
        for (int i = 0; i < 4; ++i) {
            int m = waveM * 64 + i * 16 + (lane & 15);
            a[i].u[0] = *(const v4u*)&as[m * LDA + kl];
            a[i].u[1] = *(const v4u*)&as[m * LDA + kl + 16];
        }
        for (int j = 0; j < 2; ++j) {
            int n = waveN * 32 + j * 16 + (lane & 15);
            b[j].u[0] = *(const v4u*)&bh[n * LDA + kl];
            b[j].u[1] = *(const v4u*)&bh[n * LDA + kl + 16];
        }
        for (int i = 0; i < 4; ++i)
            for (int j = 0; j < 2; ++j)
                accH[i][j] = __builtin_amdgcn_wmma_f32_16x16x32_bf16(
                    false, a[i].v, false, b[j].v, (short)0, accH[i][j], false, false);
        for (int j = 0; j < 2; ++j) {
            int n = waveN * 32 + j * 16 + (lane & 15);
            b[j].u[0] = *(const v4u*)&bg[n * LDA + kl];
            b[j].u[1] = *(const v4u*)&bg[n * LDA + kl + 16];
        }
        for (int i = 0; i < 4; ++i)
            for (int j = 0; j < 2; ++j)
                accG[i][j] = __builtin_amdgcn_wmma_f32_16x16x32_bf16(
                    false, a[i].v, false, b[j].v, (short)0, accG[i][j], false, false);
        __syncthreads();
    }

    const int halfsel = (lane >> 4) * 8;
    for (int i = 0; i < 4; ++i)
        for (int j = 0; j < 2; ++j) {
            int n = n0 + waveN * 32 + j * 16 + (lane & 15);
            for (int v = 0; v < 8; ++v) {
                int m = m0 + waveM * 64 + i * 16 + halfsel + v;
                float hh = accH[i][j][v];
                float g  = accG[i][j][v];
                float sg = g / (1.f + __expf(-g));   // silu
                Y[(size_t)m * N + n] = (__bf16)(sg * hh);
            }
        }
}

// ---------------------------------------------------------------------------
// Routing: p[t] = clamp((1 - cos(q[t-1], k[t])) / 2), boundary flag b[t]
// ---------------------------------------------------------------------------
__global__ __launch_bounds__(256)
void routing_k(const float* __restrict__ q, const float* __restrict__ k,
               const int* __restrict__ cu, int ncu,
               float* __restrict__ p, int* __restrict__ bflag, int d)
{
    __shared__ float s0[256], s1[256], s2[256];
    const int t = blockIdx.x;
    float pt;
    if (t == 0) {
        pt = 1.f;
    } else {
        const float* qr = q + (size_t)(t - 1) * d;
        const float* kr = k + (size_t)t * d;
        float dot = 0.f, nq = 0.f, nk = 0.f;
        for (int i = threadIdx.x; i < d; i += 256) {
            float qa = qr[i], kb = kr[i];
            dot += qa * kb; nq += qa * qa; nk += kb * kb;
        }
        s0[threadIdx.x] = dot; s1[threadIdx.x] = nq; s2[threadIdx.x] = nk;
        __syncthreads();
        for (int st = 128; st > 0; st >>= 1) {
            if (threadIdx.x < st) {
                s0[threadIdx.x] += s0[threadIdx.x + st];
                s1[threadIdx.x] += s1[threadIdx.x + st];
                s2[threadIdx.x] += s2[threadIdx.x + st];
            }
            __syncthreads();
        }
        float cosv = s0[0] * rsqrtf(s1[0] * s2[0]);
        pt = (1.f - cosv) * 0.5f;
    }
    bool start = false;
    for (int s = 0; s < ncu; ++s) if (cu[s] == t) start = true;
    if (start) pt = 1.f;
    pt = fminf(fmaxf(pt, 1e-4f), 1.f - 1e-4f);
    if (threadIdx.x == 0) { p[t] = pt; bflag[t] = (pt >= 0.5f) ? 1 : 0; }
}

// ---------------------------------------------------------------------------
// Segmented DeChunk + EMA scan. Per-sequence, per-column recurrence:
//   b[t]: h = (start?0:(1-p)) * h + p*z[t];  else carry.  out[t]=h
// ---------------------------------------------------------------------------
__global__ __launch_bounds__(256)
void scan_k(const float* __restrict__ z, const float* __restrict__ p,
            const int* __restrict__ bflag, const int* __restrict__ cu,
            float* __restrict__ out, int d, int blocksPerSeq)
{
    const int s  = blockIdx.x / blocksPerSeq;
    const int cb = blockIdx.x % blocksPerSeq;
    const int j  = cb * blockDim.x + threadIdx.x;
    const int t0 = cu[s], t1 = cu[s + 1];
    float h = 0.f;
    for (int t = t0; t < t1; ++t) {
        if (bflag[t]) {
            float pt = p[t];
            float a  = (t == t0) ? 0.f : (1.f - pt);
            h = a * h + pt * z[(size_t)t * d + j];
        }
        out[(size_t)t * d + j] = h;
    }
}

// ---------------------------------------------------------------------------
extern "C" void kernel_launch(void* const* d_in, const int* in_sizes, int n_in,
                              void* d_out, int out_size, void* d_ws, size_t ws_size,
                              hipStream_t stream) {
    const float* x   = (const float*)d_in[0];
    const float* Wq  = (const float*)d_in[1];
    const float* Wk  = (const float*)d_in[2];
    const float* fc1 = (const float*)d_in[3];
    const float* fc2 = (const float*)d_in[4];
    const float* nw  = (const float*)d_in[5];
    const int*   cu  = (const int*)d_in[6];
    const int ncu = in_sizes[6];

    const int D = in_sizes[5];            // 2048
    const int T = in_sizes[0] / D;        // 8192
    const int H = in_sizes[4] / D;        // 8192 (fc2 is [H, D])

    char* ws = (char*)d_ws;
    size_t off = 0;
    auto alloc = [&](size_t bytes) { size_t o = off; off += (bytes + 255) & ~(size_t)255; return o; };

    __bf16* x_bf  = (__bf16*)(ws + alloc((size_t)T * D * 2));
    __bf16* xn_bf = (__bf16*)(ws + alloc((size_t)T * D * 2));
    __bf16* wqT   = (__bf16*)(ws + alloc((size_t)D * D * 2));
    __bf16* wkT   = (__bf16*)(ws + alloc((size_t)D * D * 2));
    __bf16* fc1T  = (__bf16*)(ws + alloc((size_t)2 * H * D * 2)); // 64MB; reused as z (f32, T*D*4 = 64MB)
    __bf16* fc2T  = (__bf16*)(ws + alloc((size_t)H * D * 2));
    float*  qf    = (float*)(ws + alloc((size_t)T * D * 4));      // contiguous with kf
    float*  kf    = (float*)(ws + alloc((size_t)T * D * 4));
    float*  pbuf  = (float*)(ws + alloc((size_t)T * 4));
    int*    bbuf  = (int*)  (ws + alloc((size_t)T * 4));
    __bf16* y_bf  = (__bf16*)qf;    // alias: q/k dead after routing (T*H*2 = 128MB = q+k)
    float*  zf    = (float*)fc1T;   // alias: fc1T dead after SwiGLU GEMM

    dim3 blk(256);

    // 1) weight transpose + bf16 convert
    transpose_cvt_k<<<dim3(D / 64, D / 64), blk, 0, stream>>>(Wq, wqT, D, D);
    transpose_cvt_k<<<dim3(D / 64, D / 64), blk, 0, stream>>>(Wk, wkT, D, D);
    transpose_cvt_k<<<dim3(D / 64, (2 * H) / 64), blk, 0, stream>>>(fc1, fc1T, D, 2 * H);
    transpose_cvt_k<<<dim3(H / 64, D / 64), blk, 0, stream>>>(fc2, fc2T, H, D);

    // 2) RMSNorm + x->bf16
    rmsnorm_k<<<dim3(T), blk, 0, stream>>>(x, nw, x_bf, xn_bf, D);

    // 3) q = x @ Wq ; k = x @ Wk
    gemm_bf16_k<0><<<dim3(D / BN, T / BM), blk, 0, stream>>>(x_bf, wqT, qf, nullptr, T, D, D);
    gemm_bf16_k<0><<<dim3(D / BN, T / BM), blk, 0, stream>>>(x_bf, wkT, kf, nullptr, T, D, D);

    // 4) routing probabilities / boundaries (before y_bf overwrites q/k)
    routing_k<<<dim3(T), blk, 0, stream>>>(qf, kf, cu, ncu, pbuf, bbuf, D);

    // 5) y = silu(xn@fc1_g) * (xn@fc1_h)   (fused, bf16 out)
    gemm_swiglu_k<<<dim3(H / BN, T / BM), blk, 0, stream>>>(
        xn_bf, fc1T, fc1T + (size_t)H * D, y_bf, T, H, D);

    // 6) z = x + y @ fc2
    gemm_bf16_k<1><<<dim3(D / BN, T / BM), blk, 0, stream>>>(y_bf, fc2T, zf, x, T, D, H);

    // 7) DeChunk + EMA scan -> d_out
    const int bps = D / 256;
    scan_k<<<dim3((ncu - 1) * bps), blk, 0, stream>>>(zf, pbuf, bbuf, cu,
                                                      (float*)d_out, D, bps);
}
